// Attention_22874995818839
// MI455X (gfx1250) — compile-verified, matrix operand order checked by reading, not run on hardware
//
#include <hip/hip_runtime.h>
#include <hip/hip_bf16.h>

// Flash-attention for the reference module, CDNA5 (gfx1250, wave32) WMMA path.
// Pipeline: bn_stats -> cvt_wo -> proj (BN+grouped conv, head-split layouts)
//           -> attn_fa (TRANSPOSED scores: S^T = K*Q^T, so online-softmax stats
//              are per-lane scalars and the P fragment is born in B-frag layout
//              -- no LDS transpose, no shuffle butterflies)
//           -> out_proj (256x256 WMMA GEMM + bias).
// Workspace requirement: ~8.6 MB.

typedef __attribute__((ext_vector_type(16))) _Float16 v16h;
typedef __attribute__((ext_vector_type(8)))  _Float16 v8h;
typedef __attribute__((ext_vector_type(8)))  float    v8f;

#define BB 2
#define CC 256
#define NN 2048
#define HH 8
#define DD 32
#define INNER (HH * DD)   // 256

// ---------------------------------------------------------------------------
// WMMA 16x32 fragment load (f16), CDNA5 A-matrix layout (ISA 7.12.2):
//   lanes 0-15 : row = lane,    K in {kb..kb+7, 16+kb..16+kb+7}, kb=0
//   lanes16-31 : row = lane-16, kb=8
// Also the B-matrix (32x16) layout when `base` points at column-major data:
// column = lane&15, K contiguous with stride `ld` between columns.
// Each lane issues two 16-byte loads (global_load_b128).
// ---------------------------------------------------------------------------
__device__ __forceinline__ v16h load_frag_k32(const _Float16* __restrict__ base, int ld) {
  int lane = threadIdx.x & 31;
  int r    = lane & 15;
  int kb   = (lane & 16) ? 8 : 0;
  const _Float16* p = base + (size_t)r * (size_t)ld + kb;
  v8h lo = *(const v8h*)(p);
  v8h hi = *(const v8h*)(p + 16);
  v16h f;
#pragma unroll
  for (int i = 0; i < 8; ++i) { f[i] = lo[i]; f[8 + i] = hi[i]; }
  return f;
}

// ---------------------------------------------------------------------------
// Kernel 1: BatchNorm batch statistics -> per-channel affine (scale, shift)
// ---------------------------------------------------------------------------
__global__ __launch_bounds__(256) void bn_stats_kernel(
    const float* __restrict__ x, const float* __restrict__ gamma,
    const float* __restrict__ beta, float* __restrict__ scale,
    float* __restrict__ shift) {
  __shared__ float rs[256], rq[256];
  int c = blockIdx.x;
  int tid = threadIdx.x;
  float s = 0.f, s2 = 0.f;
  for (int b = 0; b < BB; ++b) {
    const float* row = x + (((size_t)b * CC) + c) * NN;
    for (int i = tid; i < NN; i += 256) {
      float v = row[i];
      s += v;
      s2 = fmaf(v, v, s2);
    }
  }
  rs[tid] = s; rq[tid] = s2;
  __syncthreads();
  for (int off = 128; off > 0; off >>= 1) {
    if (tid < off) { rs[tid] += rs[tid + off]; rq[tid] += rq[tid + off]; }
    __syncthreads();
  }
  if (tid == 0) {
    float invn = 1.0f / (float)(BB * NN);
    float mean = rs[0] * invn;
    float var  = rq[0] * invn - mean * mean;
    float r    = rsqrtf(var + 1e-5f);
    float sc   = gamma[c] * r;
    scale[c] = sc;
    shift[c] = beta[c] - mean * sc;
  }
}

// ---------------------------------------------------------------------------
// Kernel 2: wo (f32, 256x256) -> f16 once (reused by every output tile)
// ---------------------------------------------------------------------------
__global__ __launch_bounds__(256) void cvt_wo_kernel(const float* __restrict__ wo,
                                                     _Float16* __restrict__ wo16) {
  int i = blockIdx.x * 256 + threadIdx.x;
  if (i < CC * INNER) wo16[i] = (_Float16)wo[i];
}

// ---------------------------------------------------------------------------
// Kernel 3: BN-normalize + grouped 1x1 conv projections.
// Emits the reference's faithful '(d h)' head split WITH the q<-k swap and the
// 1/sqrt(D) scale folded into Qh. Layouts chosen for WMMA fragment loads:
//   Qh,Kh : [b, h', n, 32]  (d' contiguous)   -> QK^T frags are 2x b128/lane
//   Vt    : [b, h', 32, n]  (n contiguous)    -> P@V  frags are 2x b128/lane
// ---------------------------------------------------------------------------
__global__ __launch_bounds__(256) void proj_kernel(
    const float* __restrict__ x, const float* __restrict__ scale,
    const float* __restrict__ shift, const float* __restrict__ wk,
    const float* __restrict__ wq, const float* __restrict__ wv,
    _Float16* __restrict__ Qh, _Float16* __restrict__ Kh,
    _Float16* __restrict__ Vt) {
  __shared__ float wkS[32 * 32], wqS[32 * 32], wvS[32 * 32];
  __shared__ float scS[32], shS[32];
  const int ntile_cnt = NN / 256;
  int nt = blockIdx.x % ntile_cnt;
  int bg = blockIdx.x / ntile_cnt;
  int g  = bg % HH;
  int b  = bg / HH;
  for (int i = threadIdx.x; i < 32 * 32; i += 256) {
    wkS[i] = wk[g * 1024 + i];
    wqS[i] = wq[g * 1024 + i];
    wvS[i] = wv[g * 1024 + i];
  }
  if (threadIdx.x < 32) {
    scS[threadIdx.x] = scale[g * 32 + threadIdx.x];
    shS[threadIdx.x] = shift[g * 32 + threadIdx.x];
  }
  __syncthreads();

  int n = nt * 256 + threadIdx.x;
  float xn[32];
#pragma unroll
  for (int c = 0; c < 32; ++c)
    xn[c] = x[(((size_t)b * CC) + g * 32 + c) * NN + n] * scS[c] + shS[c];

  const float qscale = 0.17677669529663687f;  // 1/sqrt(32)
#pragma unroll 4
  for (int d = 0; d < 32; ++d) {
    float ak = 0.f, aq = 0.f, av = 0.f;
#pragma unroll
    for (int c = 0; c < 32; ++c) {
      ak = fmaf(wkS[d * 32 + c], xn[c], ak);
      aq = fmaf(wqS[d * 32 + c], xn[c], aq);
      av = fmaf(wvS[d * 32 + c], xn[c], av);
    }
    int cc = g * 32 + d;        // projected channel index
    int hh = cc & 7;            // attention head  (channel = d'*H + h')
    int dd = cc >> 3;           // attention d'
    size_t bh = (size_t)b * HH + hh;
    Qh[(bh * NN + n) * DD + dd] = (_Float16)(ak * qscale);  // q <- k (swap), pre-scaled
    Kh[(bh * NN + n) * DD + dd] = (_Float16)aq;             // k <- q (swap)
    Vt[(bh * DD + dd) * NN + n] = (_Float16)av;             // transposed V
  }
}

// ---------------------------------------------------------------------------
// Kernel 4: flash attention with TRANSPOSED score tiles.
//   S^T = A(K-tile 16keys x 32d) x B(Q^T 32d x 16queries)
//     -> C-layout: lane owns one query column; its 8 elements are keys
//        {kb..kb+7} (kb = lane>=16 ? 8 : 0) of the strip half.
//   Softmax stats (m, l) are per-lane scalars; the only cross-lane ops are two
//   shfl_xor(16) per strip (max, sum) to merge the two key-halves of each query.
//   exp(S^T - m) lands *directly* in the B-frag layout required by
//   O^T = A(V^T 16d x 32keys) x B(P^T 32keys x 16queries): pf[i]=e(s0[i]),
//   pf[8+i]=e(s1[i]). No LDS, no transposes.
//   Epilogue: each lane holds 8 contiguous d values per accumulator ->
//   two packed b128 stores into AO[b][n][h*32+d].
// ---------------------------------------------------------------------------
__global__ __launch_bounds__(32) void attn_fa(const _Float16* __restrict__ Qh,
                                              const _Float16* __restrict__ Kh,
                                              const _Float16* __restrict__ Vt,
                                              _Float16* __restrict__ AO) {
  const int ntiles = NN / 16;
  int qt = blockIdx.x % ntiles;
  int bh = blockIdx.x / ntiles;  // b*HH + h
  int h  = bh % HH;
  int b  = bh / HH;
  int lane = threadIdx.x;
  int col  = lane & 15;                // this lane's query within the tile
  int kb   = (lane & 16) ? 8 : 0;      // key-half / d-half selector

  const _Float16* Qbase = Qh + (((size_t)bh * NN) + qt * 16) * DD;
  const _Float16* Kbase = Kh + ((size_t)bh * NN) * DD;
  const _Float16* Vbase = Vt + ((size_t)bh * DD) * NN;

  v16h qf = load_frag_k32(Qbase, DD);  // B-frag of Q^T, lives across the loop
  v8f o0 = {};  // O^T rows d = kb..kb+7        (cols = queries)
  v8f o1 = {};  // O^T rows d = 16+kb..16+kb+7
  float m = -3.0e30f, l = 0.f;

  for (int j0 = 0; j0 < NN; j0 += 32) {
    v16h kf0 = load_frag_k32(Kbase + (size_t)j0 * DD, DD);        // keys j0..j0+15
    v16h kf1 = load_frag_k32(Kbase + (size_t)(j0 + 16) * DD, DD); // keys j0+16..j0+31
    v8f zero = {};
    v8f s0 = __builtin_amdgcn_wmma_f32_16x16x32_f16(false, kf0, false, qf,
                                                    (short)0, zero, false, false);
    v8f s1 = __builtin_amdgcn_wmma_f32_16x16x32_f16(false, kf1, false, qf,
                                                    (short)0, zero, false, false);
    // --- per-lane (per-query) online softmax over this 32-key strip ---
    float mx = fmaxf(s0[0], s1[0]);
#pragma unroll
    for (int p = 1; p < 8; ++p) mx = fmaxf(mx, fmaxf(s0[p], s1[p]));
    mx = fmaxf(mx, __shfl_xor(mx, 16, 32));    // merge the two key-halves
    float mnew  = fmaxf(m, mx);
    float alpha = __expf(m - mnew);
    v16h pf;
    float rs = 0.f;
#pragma unroll
    for (int p = 0; p < 8; ++p) {
      float e0 = __expf(s0[p] - mnew);
      float e1 = __expf(s1[p] - mnew);
      rs += e0 + e1;
      pf[p]     = (_Float16)e0;   // B-frag element p   <-> key kb+p      of strip
      pf[8 + p] = (_Float16)e1;   // B-frag element 8+p <-> key 16+kb+p   of strip
    }
    rs += __shfl_xor(rs, 16, 32);
    l = l * alpha + rs;
    m = mnew;
#pragma unroll
    for (int p = 0; p < 8; ++p) { o0[p] *= alpha; o1[p] *= alpha; }
    // --- O^T += V^T x P^T ---
    v16h vf0 = load_frag_k32(Vbase + j0, NN);                     // d = 0..15
    v16h vf1 = load_frag_k32(Vbase + (size_t)16 * NN + j0, NN);   // d = 16..31
    o0 = __builtin_amdgcn_wmma_f32_16x16x32_f16(false, vf0, false, pf,
                                                (short)0, o0, false, false);
    o1 = __builtin_amdgcn_wmma_f32_16x16x32_f16(false, vf1, false, pf,
                                                (short)0, o1, false, false);
  }

  // --- epilogue: normalize, pack, two b128 stores per lane ---
  float inv = 1.0f / l;
  int n = qt * 16 + col;
  size_t off = (((size_t)b * NN) + n) * INNER + h * DD;
  v8h r0, r1;
#pragma unroll
  for (int p = 0; p < 8; ++p) {
    r0[p] = (_Float16)(o0[p] * inv);   // d = kb+p
    r1[p] = (_Float16)(o1[p] * inv);   // d = 16+kb+p
  }
  *(v8h*)(AO + off + kb)      = r0;
  *(v8h*)(AO + off + 16 + kb) = r1;
}

// ---------------------------------------------------------------------------
// Kernel 5: output projection  out[b][o][n] = sum_c wo[o][c]*AO[b][n][c] + bo[o]
// One wave per 16x16 output tile, K=256 in 8 WMMA steps.
// ---------------------------------------------------------------------------
__global__ __launch_bounds__(32) void out_proj(const _Float16* __restrict__ wo16,
                                               const _Float16* __restrict__ AO,
                                               const float* __restrict__ bo,
                                               float* __restrict__ out) {
  const int otiles = CC / 16;
  const int ntiles = NN / 16;
  int ot = blockIdx.x % otiles;
  int rest = blockIdx.x / otiles;
  int nt = rest % ntiles;
  int b  = rest / ntiles;
  int lane = threadIdx.x;
  int col  = lane & 15;
  int hi8  = (lane & 16) ? 8 : 0;

  const _Float16* Abase = wo16 + (size_t)ot * 16 * INNER;            // rows = o
  const _Float16* Bbase = AO + (((size_t)b * NN) + nt * 16) * INNER; // cols = n
  v8f acc = {};
#pragma unroll
  for (int k = 0; k < INNER; k += 32) {
    v16h af = load_frag_k32(Abase + k, INNER);
    v16h bf = load_frag_k32(Bbase + k, INNER);
    acc = __builtin_amdgcn_wmma_f32_16x16x32_f16(false, af, false, bf,
                                                 (short)0, acc, false, false);
  }
#pragma unroll
  for (int p = 0; p < 8; ++p) {
    int o = ot * 16 + p + hi8;
    int n = nt * 16 + col;
    out[(((size_t)b * CC) + o) * NN + n] = acc[p] + bo[o];
  }
}

// ---------------------------------------------------------------------------
extern "C" void kernel_launch(void* const* d_in, const int* in_sizes, int n_in,
                              void* d_out, int out_size, void* d_ws, size_t ws_size,
                              hipStream_t stream) {
  const float* x     = (const float*)d_in[0];
  const float* gamma = (const float*)d_in[1];
  const float* beta  = (const float*)d_in[2];
  const float* wk    = (const float*)d_in[3];
  const float* wq    = (const float*)d_in[4];
  const float* wv    = (const float*)d_in[5];
  const float* wo    = (const float*)d_in[6];
  const float* bo    = (const float*)d_in[7];
  float* out = (float*)d_out;

  char* ws = (char*)d_ws;
  size_t off = 0;
  float* scale = (float*)(ws + off); off += CC * sizeof(float);
  float* shift = (float*)(ws + off); off += CC * sizeof(float);
  off = (off + 255) & ~(size_t)255;
  _Float16* wo16 = (_Float16*)(ws + off); off += (size_t)CC * INNER * 2;        // 128 KB
  _Float16* Qh   = (_Float16*)(ws + off); off += (size_t)BB * HH * NN * DD * 2; // 2 MB
  _Float16* Kh   = (_Float16*)(ws + off); off += (size_t)BB * HH * NN * DD * 2; // 2 MB
  _Float16* Vt   = (_Float16*)(ws + off); off += (size_t)BB * HH * NN * DD * 2; // 2 MB
  _Float16* AO   = (_Float16*)(ws + off); off += (size_t)BB * NN * INNER * 2;   // 2 MB

  bn_stats_kernel<<<dim3(CC), dim3(256), 0, stream>>>(x, gamma, beta, scale, shift);
  cvt_wo_kernel<<<dim3((CC * INNER + 255) / 256), dim3(256), 0, stream>>>(wo, wo16);
  proj_kernel<<<dim3(BB * HH * (NN / 256)), dim3(256), 0, stream>>>(
      x, scale, shift, wk, wq, wv, Qh, Kh, Vt);
  attn_fa<<<dim3(BB * HH * (NN / 16)), dim3(32), 0, stream>>>(Qh, Kh, Vt, AO);
  out_proj<<<dim3(BB * (NN / 16) * (CC / 16)), dim3(32), 0, stream>>>(wo16, AO, bo, out);
}
